// TriangleAttentionStartingNode_63531156242732
// MI455X (gfx1250) — matrix operand order repaired and checked by reading, not compiled, not run on hardware
//
#include <hip/hip_runtime.h>
#include <hip/hip_bf16.h>
#include <math.h>

// ---------------------------------------------------------------------------
// Triangle attention (starting node): B=1, N=256, D=128, H=4, C=32
//   K0 prep_weights : f32 weights -> f16, pre-swizzled into WMMA B-fragment
//                     order (Wq scaled by 1/sqrt(C)); Wo likewise
//   K1 ln_bias      : LayerNorm(Z_raw) -> Zh (f16), tb[h][q*256+k] (f32)
//   K2 proj_gemm    : [Q|K|V|G] = Zh @ Wh (WMMA, 4 N-tiles/wave for A reuse);
//                     V stored transposed Vt[(i,h,c)][k]; G=sigmoid(.+bg)
//   K3 attn         : per (i,h,qtile16): S^T = K Q^T (WMMA) -> online softmax
//                     (P lands directly in A-fragment layout, no LDS) ->
//                     O += P V (WMMA); gate + store via LDS transpose
//   K4 out_gemm     : out = Z_raw + (O*g) @ Wo + out_bias  (WMMA, 4 tiles/wave)
// ---------------------------------------------------------------------------

typedef __attribute__((ext_vector_type(16))) _Float16 v16h;
typedef __attribute__((ext_vector_type(8)))  _Float16 v8h;
typedef __attribute__((ext_vector_type(4)))  _Float16 v4h;
typedef __attribute__((ext_vector_type(8)))  float    v8f;

#define NSEQ 256
#define DMODEL 128
#define NHEAD 4
#define CHEAD 32
#define NROWS (NSEQ * NSEQ)          // 65536 pair rows
#define QSCALE 0.17677669529663687f  // 1/sqrt(32)

// ---- workspace layout (bytes) ---------------------------------------------
#define ZH_OFF   ((size_t)0)                       // 16 MB (reused as Oh)
#define QH_OFF   ((size_t)16 << 20)                // 16 MB
#define KH_OFF   ((size_t)32 << 20)                // 16 MB
#define VT_OFF   ((size_t)48 << 20)                // 16 MB  Vt[(i*4+h)*32+c][k]
#define GH_OFF   ((size_t)64 << 20)                // 16 MB
#define WH_OFF   ((size_t)80 << 20)                // 128 KB (fragment order)
#define WOH_OFF  (((size_t)80 << 20) + (256u<<10)) // 32 KB  (fragment order)
#define TB_OFF   (((size_t)80 << 20) + (512u<<10)) // 1 MB

// ---- WMMA fragment helpers (ISA 7.12.2 layouts, wave32) -------------------

// A[16 x 32] f16 row-major, leading dim ld (halves). 2 x 16B loads per lane.
__device__ inline v16h load_a_frag(const _Float16* base, int ld) {
  const int lane = threadIdx.x & 31;
  const _Float16* p = base + (lane & 15) * ld + ((lane >> 4) * 16);
  v8h lo = *(const v8h*)(p);
  v8h hi = *(const v8h*)(p + 8);
  return __builtin_shufflevector(lo, hi, 0, 1, 2, 3, 4, 5, 6, 7,
                                 8, 9, 10, 11, 12, 13, 14, 15);
}

// B = M^T where M is [16 x 32] row-major: one 32B load per lane.
__device__ inline v16h load_bT_frag(const _Float16* base, int ld) {
  const int lane = threadIdx.x & 31;
  return *(const v16h*)(base + (lane & 15) * ld + ((lane >> 4) * 16));
}

__device__ inline v8f wmma_f16(v16h a, v16h b, v8f c) {
  return __builtin_amdgcn_wmma_f32_16x16x32_f16(false, a, false, b,
                                                (short)0, c, false, false);
}

__device__ inline float fast_sigmoid(float x) {
  return __builtin_amdgcn_rcpf(1.0f + __expf(-x));
}

// S += tb[0..7] + 1e9*(mask[0..7]-1), contiguous 32B-aligned rows
__device__ inline void add_bias8(v8f& S, const float* tb, const float* mk) {
  float4 ta = *(const float4*)tb,  tb4 = *(const float4*)(tb + 4);
  float4 ma = *(const float4*)mk,  mb4 = *(const float4*)(mk + 4);
  S[0] += ta.x + 1e9f * (ma.x - 1.0f);  S[1] += ta.y + 1e9f * (ma.y - 1.0f);
  S[2] += ta.z + 1e9f * (ma.z - 1.0f);  S[3] += ta.w + 1e9f * (ma.w - 1.0f);
  S[4] += tb4.x + 1e9f * (mb4.x - 1.0f); S[5] += tb4.y + 1e9f * (mb4.y - 1.0f);
  S[6] += tb4.z + 1e9f * (mb4.z - 1.0f); S[7] += tb4.w + 1e9f * (mb4.w - 1.0f);
}

// ---- K0: weight conversion + fragment swizzle -----------------------------
__global__ __launch_bounds__(256) void prep_weights(
    const float* __restrict__ wq, const float* __restrict__ wk,
    const float* __restrict__ wv, const float* __restrict__ wg,
    const float* __restrict__ wo, _Float16* __restrict__ WhF,
    _Float16* __restrict__ WoF) {
  const int idx = blockIdx.x * blockDim.x + threadIdx.x;  // 0..65535
  {
    const int e = idx & 15;
    const int lane = (idx >> 4) & 31;
    const int tile = idx >> 9;          // kc*32 + nt
    const int kc = tile >> 5, nt = tile & 31;
    const int k = kc * 32 + (lane >> 4) * 16 + e;
    const int col = nt * 16 + (lane & 15);
    const int which = col >> 7, ee = col & 127;
    const float* w = (which == 0) ? wq : (which == 1) ? wk
                   : (which == 2) ? wv : wg;
    float val = w[k * DMODEL + ee];
    if (which == 0) val *= QSCALE;
    WhF[idx] = (_Float16)val;
  }
  if (idx < 16384) {
    const int e = idx & 15;
    const int lane = (idx >> 4) & 31;
    const int tile = idx >> 9;          // kc*8 + nt
    const int kc = tile >> 3, nt = tile & 7;
    const int k = kc * 32 + (lane >> 4) * 16 + e;
    const int col = nt * 16 + (lane & 15);
    WoF[idx] = (_Float16)wo[k * DMODEL + col];
  }
}

// ---- K1: LayerNorm + triangle-bias projection -----------------------------
__global__ __launch_bounds__(256) void ln_bias(
    const float* __restrict__ Zraw, const float* __restrict__ lnw,
    const float* __restrict__ lnb, const float* __restrict__ bpw,
    _Float16* __restrict__ Zh, float* __restrict__ Tb) {
  const int row  = blockIdx.x * (blockDim.x >> 5) + (threadIdx.x >> 5);
  const int lane = threadIdx.x & 31;
  const float4 xv = ((const float4*)(Zraw + (size_t)row * DMODEL))[lane];

  float s = xv.x + xv.y + xv.z + xv.w;
#pragma unroll
  for (int m = 1; m <= 16; m <<= 1) s += __shfl_xor(s, m, 32);
  const float mu = s * (1.0f / DMODEL);

  const float d0 = xv.x - mu, d1 = xv.y - mu, d2 = xv.z - mu, d3 = xv.w - mu;
  float ss = d0 * d0 + d1 * d1 + d2 * d2 + d3 * d3;
#pragma unroll
  for (int m = 1; m <= 16; m <<= 1) ss += __shfl_xor(ss, m, 32);
  const float rinv = rsqrtf(ss * (1.0f / DMODEL) + 1e-5f);

  const int c = lane * 4;
  const float4 wv4 = *(const float4*)(lnw + c);
  const float4 bv4 = *(const float4*)(lnb + c);
  const float z0 = d0 * rinv * wv4.x + bv4.x;
  const float z1 = d1 * rinv * wv4.y + bv4.y;
  const float z2 = d2 * rinv * wv4.z + bv4.z;
  const float z3 = d3 * rinv * wv4.w + bv4.w;

  v4h zo; zo[0] = (_Float16)z0; zo[1] = (_Float16)z1;
          zo[2] = (_Float16)z2; zo[3] = (_Float16)z3;
  *(v4h*)(Zh + (size_t)row * DMODEL + c) = zo;

#pragma unroll
  for (int h = 0; h < NHEAD; ++h) {
    const float4 bp = *(const float4*)(bpw + h * DMODEL + c);
    float t = z0 * bp.x + z1 * bp.y + z2 * bp.z + z3 * bp.w;
#pragma unroll
    for (int m = 1; m <= 16; m <<= 1) t += __shfl_xor(t, m, 32);
    if (lane == 0) Tb[(size_t)h * NROWS + row] = t;
  }
}

// ---- K2: fused Q/K/V/Gate projection GEMM ---------------------------------
// Each wave: one 16-row M tile x 4 consecutive 16-col N tiles (A reused 4x).
__global__ __launch_bounds__(256) void proj_gemm(
    const _Float16* __restrict__ Zh, const _Float16* __restrict__ WhF,
    const float* __restrict__ bg, _Float16* __restrict__ Qh,
    _Float16* __restrict__ Kh, _Float16* __restrict__ Gh,
    _Float16* __restrict__ Vt) {
  __shared__ _Float16 tbuf[8][16 * 64];   // per-wave 16x64 transpose staging
  const int wslot = threadIdx.x >> 5;
  const int lane  = threadIdx.x & 31;
  const int wave  = blockIdx.x * 8 + wslot;   // 32768 waves
  const int mt  = wave >> 3;                  // 4096 row tiles
  const int ntg = wave & 7;                   // group of 4 N-tiles

  v8f acc[4] = {};
#pragma unroll
  for (int kc = 0; kc < 4; ++kc) {
    v16h a = load_a_frag(Zh + (size_t)(mt * 16) * DMODEL + kc * 32, DMODEL);
#pragma unroll
    for (int j = 0; j < 4; ++j) {
      v16h b = *(const v16h*)(WhF +
                ((size_t)((kc * 32 + ntg * 4 + j) * 32) + lane) * 16);
      acc[j] = wmma_f16(a, b, acc[j]);
    }
  }

  const int n = lane & 15, half = lane >> 4;
  const int which = ntg >> 1;   // scalar: 0=Q 1=K 2=V 3=G

  if (which == 2) {
    // V transposed: Vt[((i*4+h)*32+c)*256 + j], 16B vector store per tile
    const int i  = mt >> 4;
    const int j0 = ((mt * 16) & 255) + half * 8;
#pragma unroll
    for (int j = 0; j < 4; ++j) {
      const int e  = ((ntg & 1) * 4 + j) * 16 + n;   // 0..127 within V block
      const int hh = e >> 5, cc = e & 31;
      v8h vv;
#pragma unroll
      for (int r = 0; r < 8; ++r) vv[r] = (_Float16)acc[j][r];
      *(v8h*)(Vt + ((size_t)((i * NHEAD + hh) * CHEAD + cc)) * NSEQ + j0) = vv;
    }
  } else {
    _Float16* tb = tbuf[wslot];
#pragma unroll
    for (int j = 0; j < 4; ++j) {
      const int e = ((ntg & 1) * 4 + j) * 16 + n;
      const float bge = (which == 3) ? bg[e] : 0.0f;
#pragma unroll
      for (int r = 0; r < 8; ++r) {
        float v = acc[j][r];
        if (which == 3) v = fast_sigmoid(v + bge);   // gate
        tb[(half * 8 + r) * 64 + j * 16 + n] = (_Float16)v;
      }
    }
    // same-wave DS ops are in-order: read back row-major, 2x32B store per lane
    _Float16* dst = (which == 0) ? Qh : (which == 1) ? Kh : Gh;
    const int row = n;
    const int cb  = (ntg & 1) * 64 + half * 32;
    v16h oa = *(const v16h*)(&tb[row * 64 + half * 32]);
    v16h ob = *(const v16h*)(&tb[row * 64 + half * 32 + 16]);
    _Float16* dp = dst + (size_t)(mt * 16 + row) * DMODEL + cb;
    *(v16h*)(dp)      = oa;
    *(v16h*)(dp + 16) = ob;
  }
}

// ---- K3: flash attention per (i, h, q-tile of 16) -------------------------
// S^T = K Q^T so P lands directly in the A-fragment layout for P @ V.
__global__ __launch_bounds__(128) void attn(
    const _Float16* __restrict__ Qh, const _Float16* __restrict__ Kh,
    const _Float16* __restrict__ Vt, const float* __restrict__ Tb,
    const float* __restrict__ mask, const _Float16* __restrict__ Gh,
    _Float16* __restrict__ Oh) {
  __shared__ float obuf[4][16 * 32];   // per-wave O-tile transpose staging
  const int wslot = threadIdx.x >> 5;
  const int lane = threadIdx.x & 31;
  const int task = blockIdx.x * 4 + wslot;   // 16384 tasks
  const int qt = task & 15;
  const int h  = (task >> 4) & 3;
  const int i  = task >> 6;
  const int q0 = qt * 16;
  const int n = lane & 15, half = lane >> 4;

  // B = Q^T fragment [32c x 16q], loaded once (32B per lane)
  const v16h bq =
      load_bT_frag(Qh + ((size_t)(i * NSEQ + q0)) * DMODEL + h * CHEAD, DMODEL);

  float mq = -1e30f, lq = 0.0f;   // running max / sum for q = n (this lane)
  v8f O0 = {}, O1 = {};
  const _Float16* Vbase = Vt + ((size_t)(i * NHEAD + h) * CHEAD) * NSEQ;
  const float* Tbq = Tb + (size_t)h * NROWS + (size_t)(q0 + n) * NSEQ;
  const float* Mki = mask + i * NSEQ;

  for (int kt = 0; kt < 8; ++kt) {
    const int k0 = kt * 32;
    // S^T tiles: rows k, cols q.  ST0[r] = S[q=n][k0+half*8+r], ST1: +16
    v16h ak0 = load_a_frag(Kh + ((size_t)(i * NSEQ + k0)) * DMODEL + h * CHEAD, DMODEL);
    v16h ak1 = load_a_frag(Kh + ((size_t)(i * NSEQ + k0 + 16)) * DMODEL + h * CHEAD, DMODEL);
    v8f ST0 = {}, ST1 = {};
    ST0 = wmma_f16(ak0, bq, ST0);
    ST1 = wmma_f16(ak1, bq, ST1);

    const int kb = k0 + half * 8;
    add_bias8(ST0, Tbq + kb,      Mki + kb);
    add_bias8(ST1, Tbq + kb + 16, Mki + kb + 16);

    // online softmax: one (m,l) per lane (its q), reduce with partner lane^16
    float tmax = -1e30f;
#pragma unroll
    for (int r = 0; r < 8; ++r) tmax = fmaxf(tmax, fmaxf(ST0[r], ST1[r]));
    tmax = fmaxf(tmax, __shfl_xor(tmax, 16, 32));
    const float mnew  = fmaxf(mq, tmax);
    const float alpha = __expf(mq - mnew);
    mq = mnew;

    v16h ap;                       // P already in A-fragment layout
    float s = 0.0f;
#pragma unroll
    for (int r = 0; r < 8; ++r) {
      const float p0 = __expf(ST0[r] - mnew);
      const float p1 = __expf(ST1[r] - mnew);
      s += p0 + p1;
      ap[r]     = (_Float16)p0;
      ap[8 + r] = (_Float16)p1;
    }
    s += __shfl_xor(s, 16, 32);
    lq = lq * alpha + s;

    // rescale O rows (row q of O = half*8 + r; its alpha lives in lane q)
#pragma unroll
    for (int r = 0; r < 8; ++r) {
      const float ar = __shfl(alpha, half * 8 + r, 32);
      O0[r] *= ar;
      O1[r] *= ar;
    }

    // O += P @ V : B fragments are 2x16B loads (k0+half*8, k0+16+half*8) from transposed V
    const _Float16* pv0 = Vbase + (size_t)n * NSEQ + k0 + half * 8;
    const _Float16* pv1 = Vbase + (size_t)(16 + n) * NSEQ + k0 + half * 8;
    v16h bv0 = __builtin_shufflevector(*(const v8h*)pv0, *(const v8h*)(pv0 + 16), 0, 1, 2, 3, 4, 5, 6, 7, 8, 9, 10, 11, 12, 13, 14, 15);
    v16h bv1 = __builtin_shufflevector(*(const v8h*)pv1, *(const v8h*)(pv1 + 16), 0, 1, 2, 3, 4, 5, 6, 7, 8, 9, 10, 11, 12, 13, 14, 15);
    O0 = wmma_f16(ap, bv0, O0);
    O1 = wmma_f16(ap, bv1, O1);
  }

  // finalize via LDS transpose: contiguous 32B gate load + 32B O store / lane
  float* ob = obuf[wslot];
#pragma unroll
  for (int r = 0; r < 8; ++r) {
    ob[(half * 8 + r) * 32 + n]      = O0[r];
    ob[(half * 8 + r) * 32 + 16 + n] = O1[r];
  }
  const int row = n;   // this lane now owns q-row `row`, c-half `half`
  const float linv = __builtin_amdgcn_rcpf(__shfl(lq, row, 32));
  const size_t rowoff =
      ((size_t)(i * NSEQ + q0 + row)) * DMODEL + h * CHEAD + half * 16;
  const v16h gv = *(const v16h*)(Gh + rowoff);
  v16h ov;
#pragma unroll
  for (int e = 0; e < 16; ++e) {
    const float oval = ob[row * 32 + half * 16 + e];
    ov[e] = (_Float16)(oval * linv * (float)gv[e]);
  }
  *(v16h*)(Oh + rowoff) = ov;
}

// ---- K4: output projection + residual -------------------------------------
// Each wave: one M tile x 4 N tiles (A reused 4x); LDS-transposed epilogue.
__global__ __launch_bounds__(256) void out_gemm(
    const _Float16* __restrict__ Oh, const _Float16* __restrict__ WoF,
    const float* __restrict__ Zraw, const float* __restrict__ out_bias,
    float* __restrict__ out) {
  __shared__ float obuf[8][16 * 64];
  const int wslot = threadIdx.x >> 5;
  const int lane  = threadIdx.x & 31;
  const int wave  = blockIdx.x * 8 + wslot;   // 8192 waves
  const int mt  = wave >> 1;                  // 4096 row tiles
  const int ntg = wave & 1;                   // group of 4 N tiles

  v8f acc[4] = {};
#pragma unroll
  for (int kc = 0; kc < 4; ++kc) {
    v16h a = load_a_frag(Oh + (size_t)(mt * 16) * DMODEL + kc * 32, DMODEL);
#pragma unroll
    for (int j = 0; j < 4; ++j) {
      v16h b = *(const v16h*)(WoF +
                ((size_t)((kc * 8 + ntg * 4 + j) * 32) + lane) * 16);
      acc[j] = wmma_f16(a, b, acc[j]);
    }
  }

  const int n = lane & 15, half = lane >> 4;
  float* ob = obuf[wslot];
#pragma unroll
  for (int j = 0; j < 4; ++j)
#pragma unroll
    for (int r = 0; r < 8; ++r)
      ob[(half * 8 + r) * 64 + j * 16 + n] = acc[j][r];

  // read back transposed (same-wave DS in order): 32 contiguous floats/lane
  const int row = n;
  const int cb  = ntg * 64 + half * 32;
  const size_t idx = (size_t)(mt * 16 + row) * DMODEL + cb;
#pragma unroll
  for (int t = 0; t < 8; ++t) {
    const float4 av = *(const float4*)(&ob[row * 64 + half * 32 + t * 4]);
    const float4 zv = *(const float4*)(Zraw + idx + t * 4);
    const float4 bv = *(const float4*)(out_bias + cb + t * 4);
    float4 o = make_float4(zv.x + av.x + bv.x, zv.y + av.y + bv.y,
                           zv.z + av.z + bv.z, zv.w + av.w + bv.w);
    *(float4*)(out + idx + t * 4) = o;
  }
}

// ---------------------------------------------------------------------------
extern "C" void kernel_launch(void* const* d_in, const int* in_sizes, int n_in,
                              void* d_out, int out_size, void* d_ws, size_t ws_size,
                              hipStream_t stream) {
  const float* Zraw  = (const float*)d_in[0];
  const float* Zmask = (const float*)d_in[1];
  const float* ln_w  = (const float*)d_in[2];
  const float* ln_b  = (const float*)d_in[3];
  const float* bpw   = (const float*)d_in[4];
  const float* wq    = (const float*)d_in[5];
  const float* wk    = (const float*)d_in[6];
  const float* wv    = (const float*)d_in[7];
  const float* wg    = (const float*)d_in[8];
  const float* bg    = (const float*)d_in[9];
  const float* wo    = (const float*)d_in[10];
  const float* obias = (const float*)d_in[11];
  float* out = (float*)d_out;

  char* ws = (char*)d_ws;
  _Float16* Zh  = (_Float16*)(ws + ZH_OFF);
  _Float16* Qh  = (_Float16*)(ws + QH_OFF);
  _Float16* Kh  = (_Float16*)(ws + KH_OFF);
  _Float16* Vt  = (_Float16*)(ws + VT_OFF);
  _Float16* Gh  = (_Float16*)(ws + GH_OFF);
  _Float16* WhF = (_Float16*)(ws + WH_OFF);
  _Float16* WoF = (_Float16*)(ws + WOH_OFF);
  float*    Tb  = (float*)(ws + TB_OFF);
  _Float16* Oh  = Zh;  // Zh dead after proj_gemm; reuse for attention output

  prep_weights<<<256, 256, 0, stream>>>(wq, wk, wv, wg, wo, WhF, WoF);
  ln_bias<<<8192, 256, 0, stream>>>(Zraw, ln_w, ln_b, bpw, Zh, Tb);
  proj_gemm<<<4096, 256, 0, stream>>>(Zh, WhF, bg, Qh, Kh, Gh, Vt);
  attn<<<4096, 128, 0, stream>>>(Qh, Kh, Vt, Tb, Zmask, Gh, Oh);
  out_gemm<<<1024, 256, 0, stream>>>(Oh, WoF, Zraw, obias, out);
}
